// FAVOR_24558622999089
// MI455X (gfx1250) — matrix operand order repaired and checked by reading, not compiled, not run on hardware
//
#include <hip/hip_runtime.h>
#include <stdint.h>

typedef __attribute__((ext_vector_type(16))) _Float16 v16h;
typedef __attribute__((ext_vector_type(8)))  _Float16 v8h;
typedef __attribute__((ext_vector_type(8)))  float    v8f;

namespace {
constexpr int kBH      = 8;     // B*H heads
constexpr int kD       = 64;    // key dim
constexpr int kDV      = 64;    // value dim
constexpr int kN       = 2048;  // sequence length
constexpr int kM       = 128;   // random features
constexpr int kC       = 64;    // chunk length
constexpr int kWaves   = 4;
constexpr int kThreads = kWaves * 32;
constexpr float kScale = 0.08838834764831845f; // 1/sqrt(M)

constexpr size_t kSmemBytes =
    (size_t)(kDV * kM + kM + kC) * sizeof(float) +           // S^T, z, norm
    (size_t)(2 * kC * kD + kDV * kC) * sizeof(float) +       // Xq, Xk, V^T
    (size_t)(kM * kD + 2 * kC * kM + kM * kC + kC * kC) * sizeof(_Float16);
} // namespace

// K index inside a 16-bit A fragment (ISA 7.12.2): two 8-wide runs per half.
__device__ __forceinline__ int kmapA(int e, int g) {
  const int v = e >> 1, p = e & 1;
  return 16 * (v >> 2) + ((v & 3) << 1) + p + 8 * g;
}

__device__ __forceinline__ v8f wmma_f16(v16h a, v16h b, v8f c) {
  return __builtin_amdgcn_wmma_f32_16x16x32_f16(false, a, false, b, (short)0, c,
                                                false, false);
}

// A fragment, row-major f16: A[m][k] = s[(r0+m)*ld + k0+k]  (2x ds_load_b128)
__device__ __forceinline__ v16h load_a_f16(const _Float16* s, int ld, int r0,
                                           int k0, int lane) {
  const int m = r0 + (lane & 15);
  const int g = lane >> 4;
  v16h a;
#pragma unroll
  for (int e = 0; e < 16; ++e) a[e] = s[m * ld + k0 + kmapA(e, g)];
  return a;
}

// A fragment from row-major f32 (contiguous runs, cvt on the fly)
__device__ __forceinline__ v16h load_a_f32(const float* s, int ld, int r0,
                                           int k0, int lane) {
  const int m = r0 + (lane & 15);
  const int g = lane >> 4;
  v16h a;
#pragma unroll
  for (int e = 0; e < 16; ++e) a[e] = (_Float16)s[m * ld + k0 + kmapA(e, g)];
  return a;
}

// B fragment where s holds B^T row-major (f16): B[k][n] = s[(n0+n)*ld + k0+k]
__device__ __forceinline__ v16h load_bT_f16(const _Float16* s, int ld, int k0,
                                            int n0, int lane) {
  const int n = n0 + (lane & 15);
  const int g = lane >> 4;
  v16h b;
#pragma unroll
  for (int e = 0; e < 16; ++e) b[e] = s[n * ld + k0 + kmapA(e, g)];
  return b;
}

// Same from f32 B^T source (two 8-float runs, cvt on the fly)
__device__ __forceinline__ v16h load_bT_f32(const float* s, int ld, int k0,
                                            int n0, int lane) {
  const int n = n0 + (lane & 15);
  const int g = lane >> 4;
  v16h b;
#pragma unroll
  for (int e = 0; e < 16; ++e) b[e] = (_Float16)s[n * ld + k0 + kmapA(e, g)];
  return b;
}

__device__ __forceinline__ v8h relu_scale_pack(const v8f& a) {
  v8h h;
#pragma unroll
  for (int i = 0; i < 8; ++i) {
    const float v = a[i];
    h[i] = (_Float16)(v > 0.f ? v * kScale : 0.f);
  }
  return h;
}

// Async global -> LDS copy of one dword per lane (ASYNCcnt path)
__device__ __forceinline__ void async_copy_b32(void* lds, const float* gptr) {
  const uint32_t lds_off = (uint32_t)(uintptr_t)lds;
  const uint64_t gaddr   = (uint64_t)(uintptr_t)gptr;
  asm volatile("global_load_async_to_lds_b32 %0, %1, off"
               :: "v"(lds_off), "v"(gaddr) : "memory");
}

__device__ __forceinline__ void wait_async_all() {
  asm volatile("s_wait_asynccnt 0x0" ::: "memory");
}

__global__ __launch_bounds__(kThreads) void favor_chunked_wmma(
    const float* __restrict__ keys, const float* __restrict__ values,
    const float* __restrict__ queries, const float* __restrict__ features,
    float* __restrict__ out) {
  extern __shared__ __align__(32) char smem_raw[];
  float*    sSt    = (float*)smem_raw;            // [kDV][kM] S^T state, f32
  float*    sZ     = sSt + kDV * kM;              // [kM]
  float*    sNorm  = sZ + kM;                     // [kC]
  float*    sXq    = sNorm + kC;                  // [kC][kD]  staged Q, f32
  float*    sXk    = sXq + kC * kD;               // [kC][kD]  staged K, f32
  float*    sVt    = sXk + kC * kD;               // [kDV][kC] staged V^T, f32
  _Float16* sFt    = (_Float16*)(sVt + kDV * kC); // [kM][kD]  features (natural)
  _Float16* sPhiQ  = sFt + kM * kD;               // [kC][kM]
  _Float16* sPhiK  = sPhiQ + kC * kM;             // [kC][kM]
  _Float16* sPhiKt = sPhiK + kC * kM;             // [kM][kC]
  _Float16* sAm    = sPhiKt + kM * kC;            // [kC][kC'] masked scores

  const int head = blockIdx.x;
  const int tid  = threadIdx.x;
  const int lane = tid & 31;
  const int wave = tid >> 5;
  const int g    = lane >> 4;
  const int col  = 16 * wave + (lane & 15); // this wave's owned column index

  const float* Kh = keys    + (size_t)head * kD * kN;
  const float* Vh = values  + (size_t)head * kDV * kN;
  const float* Qh = queries + (size_t)head * kD * kN;
  float*       Oh = out     + (size_t)head * kDV * kN;

  for (int i = tid; i < kDV * kM; i += kThreads) sSt[i] = 0.f;
  for (int i = tid; i < kM; i += kThreads) sZ[i] = 0.f;
  for (int i = tid; i < kM * kD; i += kThreads)
    sFt[i] = (_Float16)features[i]; // natural [m][d] layout
  __syncthreads();

  for (int n0 = 0; n0 < kN; n0 += kC) {
    // ---- async-stage Q, K (as [c][d]) and V (as [dv][c]) tiles into LDS
    for (int i = tid; i < kC * kD; i += kThreads) {
      const int c = i % kC, d = i / kC;
      async_copy_b32(&sXq[c * kD + d], Qh + (size_t)d * kN + n0 + c);
      async_copy_b32(&sXk[c * kD + d], Kh + (size_t)d * kN + n0 + c);
      async_copy_b32(&sVt[d * kC + c], Vh + (size_t)d * kN + n0 + c);
    }
    wait_async_all();
    __syncthreads();

    // ---- phi^T pass: phi[m][c] = relu(F[m][:] . X[c][:]) * scale
    // M-dim = m (contiguous in dst row) -> packed b128 stores into [c][m].
    // Wave owns c-tile = wave; F A-fragments shared by phiQ and phiK.
    {
      v16h bq[2], bk[2];
#pragma unroll
      for (int k = 0; k < 2; ++k) {
        bq[k] = load_bT_f32(sXq, kD, 32 * k, 16 * wave, lane);
        bk[k] = load_bT_f32(sXk, kD, 32 * k, 16 * wave, lane);
      }
#pragma unroll
      for (int ti = 0; ti < 8; ++ti) {
        v16h a0 = load_a_f16(sFt, kD, 16 * ti, 0, lane);
        v16h a1 = load_a_f16(sFt, kD, 16 * ti, 32, lane);
        v8f aq = {}, ak = {};
        aq = wmma_f16(a0, bq[0], aq); aq = wmma_f16(a1, bq[1], aq);
        ak = wmma_f16(a0, bk[0], ak); ak = wmma_f16(a1, bk[1], ak);
        *(v8h*)&sPhiQ[col * kM + 16 * ti + 8 * g] = relu_scale_pack(aq);
        *(v8h*)&sPhiK[col * kM + 16 * ti + 8 * g] = relu_scale_pack(ak);
      }
    }
    // ---- second orientation for phiK^T copy: M-dim = c -> packed into [m][c]
    {
      v16h a0 = load_a_f32(sXk, kD, 16 * wave, 0, lane);
      v16h a1 = load_a_f32(sXk, kD, 16 * wave, 32, lane);
#pragma unroll
      for (int tj = 0; tj < 8; ++tj) {
        v16h b0 = load_bT_f16(sFt, kD, 0, 16 * tj, lane);
        v16h b1 = load_bT_f16(sFt, kD, 32, 16 * tj, lane);
        v8f acc = {};
        acc = wmma_f16(a0, b0, acc); acc = wmma_f16(a1, b1, acc);
        const int m = 16 * tj + (lane & 15);
        *(v8h*)&sPhiKt[m * kC + 16 * wave + 8 * g] = relu_scale_pack(acc);
      }
    }
    __syncthreads();

    // ---- scores transposed: AT[c'][c] = phiK[c'] . phiQ[c]; mask c' <= c.
    // M-dim = c' (contiguous in dst row) -> packed stores into sAm[c][c'].
    {
      v16h bq[4];
#pragma unroll
      for (int k = 0; k < 4; ++k)
        bq[k] = load_bT_f16(sPhiQ, kM, 32 * k, 16 * wave, lane);
#pragma unroll
      for (int ti = 0; ti < 4; ++ti) {
        v8f acc = {};
#pragma unroll
        for (int k = 0; k < 4; ++k)
          acc = wmma_f16(load_a_f16(sPhiK, kM, 16 * ti, 32 * k, lane), bq[k],
                         acc);
        v8h h;
#pragma unroll
        for (int i = 0; i < 8; ++i) {
          const int cp = 16 * ti + i + 8 * g; // key position c'
          h[i] = (_Float16)((cp <= col) ? acc[i] : 0.f);
        }
        *(v8h*)&sAm[col * kC + 16 * ti + 8 * g] = h;
      }
    }
    __syncthreads();

    // ---- norm[c] = rowsum(A_masked[c]) + phiQ[c] . z
    for (int c = tid; c < kC; c += kThreads) {
      float s = 0.f;
      for (int j = 0; j < kC; ++j) s += (float)sAm[c * kC + j];
      for (int m = 0; m < kM; ++m) s += (float)sPhiQ[c * kM + m] * sZ[m];
      sNorm[c] = s;
    }
    __syncthreads();

    // ---- out = (A @ V + phiQ @ S) / norm ; wave owns dv-tile = wave.
    // D-frag rows (c) contiguous in (dv, n) output -> one v8f store per tile.
    {
      v16h bv[2], bs[4];
#pragma unroll
      for (int k = 0; k < 2; ++k)
        bv[k] = load_bT_f32(sVt, kC, 32 * k, 16 * wave, lane);
#pragma unroll
      for (int k = 0; k < 4; ++k)
        bs[k] = load_bT_f32(sSt, kM, 32 * k, 16 * wave, lane);
#pragma unroll
      for (int ti = 0; ti < 4; ++ti) {
        v8f acc = {};
#pragma unroll
        for (int k = 0; k < 2; ++k)
          acc = wmma_f16(load_a_f16(sAm, kC, 16 * ti, 32 * k, lane), bv[k], acc);
#pragma unroll
        for (int k = 0; k < 4; ++k)
          acc = wmma_f16(load_a_f16(sPhiQ, kM, 16 * ti, 32 * k, lane), bs[k],
                         acc);
        const v8f vn = *(const v8f*)&sNorm[16 * ti + 8 * g];
        v8f o;
#pragma unroll
        for (int i = 0; i < 8; ++i) o[i] = acc[i] / vn[i];
        *(v8f*)&Oh[(size_t)col * kN + n0 + 16 * ti + 8 * g] = o;
      }
    }
    __syncthreads(); // all reads of sSt complete before state update

    // ---- S^T[dv][m] += (phiK^T @ V) : M-dim = m (contiguous in sSt row)
    // -> C/D fragments are contiguous v8f LDS accesses. Wave owns dv-tile.
    {
      v16h bv[2];
#pragma unroll
      for (int k = 0; k < 2; ++k)
        bv[k] = load_bT_f32(sVt, kC, 32 * k, 16 * wave, lane);
#pragma unroll
      for (int ti = 0; ti < 8; ++ti) {
        float* cptr = &sSt[col * kM + 16 * ti + 8 * g];
        v8f acc = *(const v8f*)cptr;
        acc = wmma_f16(load_a_f16(sPhiKt, kC, 16 * ti, 0, lane), bv[0], acc);
        acc = wmma_f16(load_a_f16(sPhiKt, kC, 16 * ti, 32, lane), bv[1], acc);
        *(v8f*)cptr = acc;
      }
    }
    // ---- z += colsum(phiK) (contiguous via transposed copy)
    for (int m = tid; m < kM; m += kThreads) {
      float s = sZ[m];
      for (int c = 0; c < kC; ++c) s += (float)sPhiKt[m * kC + c];
      sZ[m] = s;
    }
    __syncthreads();
  }
}

extern "C" void kernel_launch(void* const* d_in, const int* in_sizes, int n_in,
                              void* d_out, int out_size, void* d_ws,
                              size_t ws_size, hipStream_t stream) {
  (void)in_sizes; (void)n_in; (void)out_size; (void)d_ws; (void)ws_size;
  const float* keys     = (const float*)d_in[0];
  const float* values   = (const float*)d_in[1];
  const float* queries  = (const float*)d_in[2];
  const float* features = (const float*)d_in[3];
  float* out = (float*)d_out;
  favor_chunked_wmma<<<dim3(kBH), dim3(kThreads), kSmemBytes, stream>>>(
      keys, values, queries, features, out);
}
